// GPTTransformer_36043365548555
// MI455X (gfx1250) — matrix operand order repaired and checked by reading, not compiled, not run on hardware
//
#include <hip/hip_runtime.h>
#include <hip/hip_bf16.h>

// ---------------------------------------------------------------------------
// GPT-2 small forward pass for gfx1250 (MI455X).
// bf16 WMMA GEMMs with ASYNC global->LDS double-buffered staging.
// L=12 H=12 E=768 HS=64 T=1024 B=4 V=50257
// ---------------------------------------------------------------------------

typedef __bf16 bf16;
typedef __attribute__((ext_vector_type(16))) __bf16 v16bf;
typedef __attribute__((ext_vector_type(8)))  __bf16 v8bf;
typedef __attribute__((ext_vector_type(8)))  float  v8f;
typedef __attribute__((ext_vector_type(4)))  int    v4i;

#define NL 12
#define NH 12
#define NE 768
#define NHS 64
#define NT 1024
#define NB 4
#define NV 50257
#define NVPAD 50432   /* ceil(50257/256)*256 (also mult of 128) */
#define NM (NB*NT)    /* 4096 token rows */

#define KSTAGE 64
#define LPAD   72     /* 64 + 8 bf16 pad -> +16B to break bank aliasing */

typedef __attribute__((address_space(1))) v4i* gptr4;
typedef __attribute__((address_space(3))) v4i* lptr4;
typedef __attribute__((address_space(3))) void lds_void;

__device__ __forceinline__ bf16 f2bf(float f) {
  unsigned u = __builtin_bit_cast(unsigned, f);
  unsigned r = u + 0x7FFFu + ((u >> 16) & 1u);
  unsigned short h = (unsigned short)(r >> 16);
  return __builtin_bit_cast(bf16, h);
}

// 16B async copy global -> LDS (ASYNCcnt-tracked).
__device__ __forceinline__ void async_copy16(const bf16* g, bf16* l) {
#if __has_builtin(__builtin_amdgcn_global_load_async_to_lds_b128)
  __builtin_amdgcn_global_load_async_to_lds_b128((gptr4)g, (lptr4)l, 0, 0);
#else
  unsigned lds_off = (unsigned)(size_t)(lds_void*)l;
  unsigned long long ga = (unsigned long long)(size_t)g;
  asm volatile("global_load_async_to_lds_b128 %0, %1, off"
               :: "v"(lds_off), "v"(ga) : "memory");
#endif
}

__device__ __forceinline__ void wait_async_all() {
#if __has_builtin(__builtin_amdgcn_s_wait_asynccnt)
  __builtin_amdgcn_s_wait_asynccnt(0);
#else
  asm volatile("s_wait_asynccnt 0x0" ::: "memory");
#endif
}

// A fragment, 16x32 bf16: lane holds row (lane&15); lanes 0-15 carry
// K [0..15]; lanes 16-31 carry [16..31].
__device__ __forceinline__ v16bf load_a_frag(const bf16* rowptr, int kbase,
                                             int lane) {
  int sel = (lane >> 4) * 16;
  v8bf lo = *(const v8bf*)(rowptr + kbase + sel);
  v8bf hi = *(const v8bf*)(rowptr + kbase + sel + 8);
  return __builtin_shufflevector(lo, hi, 0,1,2,3,4,5,6,7,8,9,10,11,12,13,14,15);
}

// B fragment, 32x16 bf16 from Bt[n,k]: lane holds column n=(lane&15),
// contiguous K chunk at kbase + (lane>>4)*16.
__device__ __forceinline__ v16bf load_b_frag(const bf16* nrowptr, int kbase,
                                             int lane) {
  return *(const v16bf*)(nrowptr + kbase + ((lane >> 4) * 16));
}

__device__ __forceinline__ v8f wmma_bf16(v16bf a, v16bf b, v8f c) {
  return __builtin_amdgcn_wmma_f32_16x16x32_bf16(false, a, false, b,
                                                 (short)0, c, false, false);
}

// ---------------------------------------------------------------------------
// Embedding: x[b,t,:] = tok_emb[idx[b,t]] + pos_emb[t]
// ---------------------------------------------------------------------------
__global__ void embed_kernel(const int* __restrict__ idx,
                             const float* __restrict__ tok,
                             const float* __restrict__ pos,
                             float* __restrict__ x) {
  int i = blockIdx.x * 256 + threadIdx.x;
  if (i >= NM * NE) return;
  int e = i % NE; int bt = i / NE; int t = bt % NT;
  x[i] = tok[(size_t)idx[bt] * NE + e] + pos[(size_t)t * NE + e];
}

// ---------------------------------------------------------------------------
// LayerNorm: one wave per row of 768, writes bf16 normalized row.
// ---------------------------------------------------------------------------
__global__ __launch_bounds__(256) void layernorm_kernel(
    const float* __restrict__ x, const float* __restrict__ g,
    const float* __restrict__ b, bf16* __restrict__ out, int rows) {
  int lane = threadIdx.x & 31, wid = threadIdx.x >> 5;
  int row = blockIdx.x * 8 + wid;
  if (row >= rows) return;
  const float* xr = x + (size_t)row * NE;
  float v[24];
  float s = 0.f;
#pragma unroll
  for (int i = 0; i < 24; ++i) { v[i] = xr[lane + i * 32]; s += v[i]; }
#pragma unroll
  for (int off = 16; off; off >>= 1) s += __shfl_xor(s, off, 32);
  float mu = s * (1.f / NE);
  float q = 0.f;
#pragma unroll
  for (int i = 0; i < 24; ++i) { float d = v[i] - mu; q += d * d; }
#pragma unroll
  for (int off = 16; off; off >>= 1) q += __shfl_xor(q, off, 32);
  float rstd = rsqrtf(q * (1.f / NE) + 1e-8f);
#pragma unroll
  for (int i = 0; i < 24; ++i) {
    int e = lane + i * 32;
    out[(size_t)row * NE + e] = f2bf((v[i] - mu) * rstd * g[e] + b[e]);
  }
}

// ---------------------------------------------------------------------------
// Weight conversion kernels (fp32 -> bf16, into transposed Bt[n,k] layout).
// ---------------------------------------------------------------------------
__global__ void qkvw_kernel(const float* __restrict__ wq,
                            const float* __restrict__ wk,
                            const float* __restrict__ wv,
                            bf16* __restrict__ dst) {
  int i = blockIdx.x * 256 + threadIdx.x;
  if (i >= 3 * NE * NE) return;
  int e = i % NE; int n = i / NE;
  int part = n / NE; int nn = n % NE; int h = nn >> 6; int d = nn & 63;
  const float* w = (part == 0) ? wq : ((part == 1) ? wk : wv);
  dst[i] = f2bf(w[((size_t)h * NE + e) * NHS + d]);
}

// src [K,C] row-major -> dst [Cpad,K] row-major (zero-pad n >= C).
__global__ void transcvt_kernel(const float* __restrict__ src,
                                bf16* __restrict__ dst, int K, int C, int Cpad) {
  int i = blockIdx.x * 256 + threadIdx.x;
  if (i >= Cpad * K) return;
  int k = i % K; int n = i / K;
  dst[i] = f2bf(n < C ? src[(size_t)k * C + n] : 0.f);
}

// plain elementwise convert (proj_w is consumed as W[n,k] already).
__global__ void cvt_kernel(const float* __restrict__ src,
                           bf16* __restrict__ dst, int n) {
  int i = blockIdx.x * 256 + threadIdx.x;
  if (i < n) dst[i] = f2bf(src[i]);
}

// v-part of qkv [4096,2304] -> vt[b,h,d,t] (per-head transposed V).
__global__ void vtrans_kernel(const bf16* __restrict__ qkv,
                              bf16* __restrict__ vt) {
  int i = blockIdx.x * 256 + threadIdx.x;
  if (i >= NB * NH * NHS * NT) return;
  int t = i % NT; int rest = i / NT; int d = rest % NHS; int bh = rest / NHS;
  int b = bh / NH, h = bh % NH;
  vt[i] = qkv[((size_t)(b * NT + t)) * (3 * NE) + 2 * NE + h * NHS + d];
}

// ---------------------------------------------------------------------------
// bf16 WMMA GEMM with async global->LDS double-buffered staging.
// C[M,N] = A[M,K] * Bt[N,K]^T  (+bias)(gelu)(+resid)
// Block = 256 threads = 8 waves (2M x 4N); WG tile 64M x 128N; wave 32M x 32N.
// K staged in chunks of 64 through LDS.  grid.x = Npad/128, grid.y = M/64.
// flags bit0 = exact GELU.
// ---------------------------------------------------------------------------
__global__ __launch_bounds__(256) void gemm_bf16_kernel(
    const bf16* __restrict__ A, int lda,
    const bf16* __restrict__ Bt, int ldb,
    float* __restrict__ Cf, bf16* __restrict__ Cb, int ldc,
    const float* __restrict__ bias, const float* __restrict__ resid,
    int N, int K, int flags) {
  __shared__ bf16 sA[2][64][LPAD];
  __shared__ bf16 sB[2][128][LPAD];

  const int tid  = threadIdx.x;
  const int lane = tid & 31;
  const int wid  = tid >> 5;
  const int wm = wid >> 2, wn = wid & 3;
  const int mblk = blockIdx.y * 64;
  const int nblk = blockIdx.x * 128;
  const int lrow = lane & 15, lhalf = lane >> 4;

  const int crow = tid >> 3;       // 0..31 copy row
  const int cseg = (tid & 7) * 8;  // 16B segment within 64-elem K stage

  v8f acc[2][2] = {};

  const int nstage = K / KSTAGE;
  // ---- issue stage 0 copies ----
  {
    const bf16* ag = A + (size_t)(mblk + crow) * lda + cseg;
    async_copy16(ag, &sA[0][crow][cseg]);
    async_copy16(ag + (size_t)32 * lda, &sA[0][crow + 32][cseg]);
#pragma unroll
    for (int j = 0; j < 4; ++j) {
      int r = crow + j * 32;
      const bf16* bg = Bt + (size_t)(nblk + r) * ldb + cseg;
      async_copy16(bg, &sB[0][r][cseg]);
    }
  }
  wait_async_all();
  __syncthreads();

  int buf = 0;
  for (int s = 0; s < nstage; ++s) {
    // ---- prefetch next stage into the other buffer (ASYNCcnt-tracked) ----
    if (s + 1 < nstage) {
      const int k0 = (s + 1) * KSTAGE;
      const bf16* ag = A + (size_t)(mblk + crow) * lda + k0 + cseg;
      async_copy16(ag, &sA[buf ^ 1][crow][cseg]);
      async_copy16(ag + (size_t)32 * lda, &sA[buf ^ 1][crow + 32][cseg]);
#pragma unroll
      for (int j = 0; j < 4; ++j) {
        int r = crow + j * 32;
        const bf16* bg = Bt + (size_t)(nblk + r) * ldb + k0 + cseg;
        async_copy16(bg, &sB[buf ^ 1][r][cseg]);
      }
    }
    // ---- compute current stage: 2 k-chunks of 32, 4 WMMAs each ----
#pragma unroll
    for (int kc = 0; kc < 2; ++kc) {
      const int kb = kc * 32;
      v16bf a0 = load_a_frag(&sA[buf][wm * 32 + lrow][0], kb, lane);
      v16bf a1 = load_a_frag(&sA[buf][wm * 32 + 16 + lrow][0], kb, lane);
#pragma unroll
      for (int j = 0; j < 2; ++j) {
        v16bf b = load_b_frag(&sB[buf][wn * 32 + j * 16 + lrow][0], kb, lane);
        acc[0][j] = wmma_bf16(a0, b, acc[0][j]);
        acc[1][j] = wmma_bf16(a1, b, acc[1][j]);
      }
    }
    wait_async_all();
    __syncthreads();
    buf ^= 1;
  }

  // ---- epilogue: bias -> gelu -> residual, fp32 and/or bf16 stores ----
#pragma unroll
  for (int i = 0; i < 2; ++i) {
#pragma unroll
    for (int j = 0; j < 2; ++j) {
#pragma unroll
      for (int r = 0; r < 8; ++r) {
        int row = mblk + wm * 32 + i * 16 + r + 8 * lhalf;
        int col = nblk + wn * 32 + j * 16 + lrow;
        if (col < N) {
          float v = acc[i][j][r];
          if (bias) v += bias[col];
          if (flags & 1) v = 0.5f * v * (1.f + erff(v * 0.70710678118654752f));
          if (resid) v += resid[(size_t)row * ldc + col];
          if (Cf) Cf[(size_t)row * ldc + col] = v;
          if (Cb) Cb[(size_t)row * ldc + col] = f2bf(v);
        }
      }
    }
  }
}

// ---------------------------------------------------------------------------
// Flash-style causal attention. 1 wave per (16 query rows, head, batch).
// qkv: [4096, 2304] bf16 (q|k|v per head).  vt: [B,H,64,1024] bf16.
// o:   [4096, 768] bf16.  Online softmax through LDS, WMMA for S and P*V.
// ---------------------------------------------------------------------------
__global__ __launch_bounds__(32) void attn_kernel(
    const bf16* __restrict__ qkv, const bf16* __restrict__ vt,
    bf16* __restrict__ o) {
  __shared__ float sS[16][32];
  __shared__ bf16  sP[16][32];
  __shared__ float sM[16], sL[16], sA[16];

  const int lane  = threadIdx.x;
  const int mtile = blockIdx.x;   // 0..63 (16-row query tiles)
  const int h     = blockIdx.y;   // 0..11
  const int b     = blockIdx.z;   // 0..3
  const int m0    = mtile * 16;
  const int lrow  = lane & 15, lhalf = lane >> 4;
  const float scale = 0.036084391824351613f;  // 768^-0.5 (reference: E^-0.5)

  const bf16* qrow = qkv + ((size_t)(b * NT + m0 + lrow)) * (3 * NE) + h * NHS;
  v16bf aq0 = load_a_frag(qrow, 0, lane);
  v16bf aq1 = load_a_frag(qrow, 32, lane);

  v8f oacc[4] = {};
  if (lane < 16) { sM[lane] = -1e30f; sL[lane] = 0.f; }
  __syncthreads();

  const int nkt = mtile / 2 + 1;  // 32-key tiles up to the causal diagonal
  for (int kt = 0; kt < nkt; ++kt) {
    const int kbase = kt * 32;
#pragma unroll
    for (int s = 0; s < 2; ++s) {
      int key = kbase + s * 16 + lrow;
      const bf16* krow =
          qkv + ((size_t)(b * NT + key)) * (3 * NE) + NE + h * NHS;
      v16bf bk0 = load_b_frag(krow, 0, lane);
      v16bf bk1 = load_b_frag(krow, 32, lane);
      v8f sfr = {};
      sfr = wmma_bf16(aq0, bk0, sfr);
      sfr = wmma_bf16(aq1, bk1, sfr);
#pragma unroll
      for (int r = 0; r < 8; ++r)
        sS[r + 8 * lhalf][s * 16 + lrow] = sfr[r] * scale;
    }
    __syncthreads();
    if (lane < 16) {
      int grow = m0 + lane;
      float mold = sM[lane], lold = sL[lane];
      float sv[32];
      float cmax = -1e30f;
#pragma unroll
      for (int j = 0; j < 32; ++j) {
        float xv = sS[lane][j];
        if (kbase + j > grow) xv = -1e30f;  // causal mask
        sv[j] = xv;
        cmax = fmaxf(cmax, xv);
      }
      float mnew = fmaxf(mold, cmax);
      float alpha = __expf(mold - mnew);
      float psum = 0.f;
#pragma unroll
      for (int j = 0; j < 32; ++j) {
        float p = (sv[j] <= -1e29f) ? 0.f : __expf(sv[j] - mnew);
        psum += p;
        sP[lane][j] = f2bf(p);
      }
      sM[lane] = mnew;
      sL[lane] = alpha * lold + psum;
      sA[lane] = alpha;
    }
    __syncthreads();
#pragma unroll
    for (int r = 0; r < 8; ++r) {
      float a = sA[r + 8 * lhalf];
#pragma unroll
      for (int j = 0; j < 4; ++j) oacc[j][r] *= a;
    }
    v16bf pa = load_a_frag(&sP[lrow][0], 0, lane);
#pragma unroll
    for (int j = 0; j < 4; ++j) {
      const bf16* vrow =
          vt + ((size_t)((b * NH + h) * NHS + j * 16 + lrow)) * NT;
      v16bf bv = load_b_frag(vrow, kbase, lane);
      oacc[j] = wmma_bf16(pa, bv, oacc[j]);
    }
    __syncthreads();
  }
  if (lane < 16) sA[lane] = 1.f / sL[lane];
  __syncthreads();
#pragma unroll
  for (int r = 0; r < 8; ++r) {
    float linv = sA[r + 8 * lhalf];
    int row = b * NT + m0 + r + 8 * lhalf;
#pragma unroll
    for (int j = 0; j < 4; ++j)
      o[(size_t)row * NE + h * NHS + j * 16 + lrow] = f2bf(oacc[j][r] * linv);
  }
}

// ---------------------------------------------------------------------------
// Host-side orchestration.
// ---------------------------------------------------------------------------
static inline size_t alignup(size_t v) { return (v + 255) & ~(size_t)255; }

extern "C" void kernel_launch(void* const* d_in, const int* in_sizes, int n_in,
                              void* d_out, int out_size, void* d_ws,
                              size_t ws_size, hipStream_t stream) {
  (void)in_sizes; (void)n_in; (void)out_size; (void)ws_size;
  const int*   idx    = (const int*)  d_in[0];
  const float* tok    = (const float*)d_in[1];
  const float* pos    = (const float*)d_in[2];
  const float* wq     = (const float*)d_in[3];
  const float* wk     = (const float*)d_in[4];
  const float* wv     = (const float*)d_in[5];
  const float* projw  = (const float*)d_in[6];
  const float* projb  = (const float*)d_in[7];
  const float* ln1g   = (const float*)d_in[8];
  const float* ln1b   = (const float*)d_in[9];
  const float* ln2g   = (const float*)d_in[10];
  const float* ln2b   = (const float*)d_in[11];
  const float* w1     = (const float*)d_in[12];
  const float* b1     = (const float*)d_in[13];
  const float* w2     = (const float*)d_in[14];
  const float* b2     = (const float*)d_in[15];
  const float* lnfg   = (const float*)d_in[16];
  const float* lnfb   = (const float*)d_in[17];
  const float* headw  = (const float*)d_in[18];
  const float* headb  = (const float*)d_in[19];
  float* out = (float*)d_out;

  // workspace carve-up (~153 MB)
  char* ws = (char*)d_ws;
  size_t off = 0;
  float* x   = (float*)(ws + off); off += alignup((size_t)NM * NE * 4);
  bf16* xn   = (bf16*) (ws + off); off += alignup((size_t)NM * NE * 2);
  bf16* qkv  = (bf16*) (ws + off); off += alignup((size_t)NM * 3 * NE * 2);
  bf16* vt   = (bf16*) (ws + off); off += alignup((size_t)NB * NH * NHS * NT * 2);
  bf16* ob   = (bf16*) (ws + off); off += alignup((size_t)NM * NE * 2);
  bf16* hb   = (bf16*) (ws + off); off += alignup((size_t)NM * 4 * NE * 2);
  bf16* wbt  = (bf16*) (ws + off); off += alignup((size_t)NVPAD * NE * 2);

  const dim3 blk256(256), blk32(32);
  const dim3 gLN(NM / 8);

  embed_kernel<<<dim3((NM * NE + 255) / 256), blk256, 0, stream>>>(idx, tok, pos, x);

  for (int l = 0; l < NL; ++l) {
    // ---- attention block ----
    layernorm_kernel<<<gLN, blk256, 0, stream>>>(x, ln1g + l * NE, ln1b + l * NE, xn, NM);
    qkvw_kernel<<<dim3((3 * NE * NE + 255) / 256), blk256, 0, stream>>>(
        wq + (size_t)l * NH * NE * NHS, wk + (size_t)l * NH * NE * NHS,
        wv + (size_t)l * NH * NE * NHS, wbt);
    gemm_bf16_kernel<<<dim3(3 * NE / 128, NM / 64), blk256, 0, stream>>>(
        xn, NE, wbt, NE, (float*)nullptr, qkv, 3 * NE,
        (const float*)nullptr, (const float*)nullptr, 3 * NE, NE, 0);
    vtrans_kernel<<<dim3((NB * NH * NHS * NT + 255) / 256), blk256, 0, stream>>>(qkv, vt);
    attn_kernel<<<dim3(NT / 16, NH, NB), blk32, 0, stream>>>(qkv, vt, ob);
    cvt_kernel<<<dim3((NE * NE + 255) / 256), blk256, 0, stream>>>(
        projw + (size_t)l * NE * NE, wbt, NE * NE);
    gemm_bf16_kernel<<<dim3(NE / 128, NM / 64), blk256, 0, stream>>>(
        ob, NE, wbt, NE, x, (bf16*)nullptr, NE,
        projb + l * NE, x, NE, NE, 0);   // x = x + o @ projW^T + b

    // ---- MLP block ----
    layernorm_kernel<<<gLN, blk256, 0, stream>>>(x, ln2g + l * NE, ln2b + l * NE, xn, NM);
    transcvt_kernel<<<dim3((4 * NE * NE + 255) / 256), blk256, 0, stream>>>(
        w1 + (size_t)l * NE * 4 * NE, wbt, NE, 4 * NE, 4 * NE);
    gemm_bf16_kernel<<<dim3(4 * NE / 128, NM / 64), blk256, 0, stream>>>(
        xn, NE, wbt, NE, (float*)nullptr, hb, 4 * NE,
        b1 + l * 4 * NE, (const float*)nullptr, 4 * NE, NE, 1);  // GELU
    transcvt_kernel<<<dim3((4 * NE * NE + 255) / 256), blk256, 0, stream>>>(
        w2 + (size_t)l * 4 * NE * NE, wbt, 4 * NE, NE, NE);
    gemm_bf16_kernel<<<dim3(NE / 128, NM / 64), blk256, 0, stream>>>(
        hb, 4 * NE, wbt, 4 * NE, x, (bf16*)nullptr, NE,
        b2 + l * NE, x, NE, 4 * NE, 0);  // x = x + h @ w2 + b2
  }

  // ---- final LN + LM head ----
  layernorm_kernel<<<gLN, blk256, 0, stream>>>(x, lnfg, lnfb, xn, NM);
  transcvt_kernel<<<dim3(((size_t)NVPAD * NE + 255) / 256), blk256, 0, stream>>>(
      headw, wbt, NE, NV, NVPAD);
  gemm_bf16_kernel<<<dim3(NVPAD / 128, NM / 64), blk256, 0, stream>>>(
      xn, NE, wbt, NE, out, (bf16*)nullptr, NV,
      headb, (const float*)nullptr, NV, NE, 0);
}
